// GPT2LMHeadModel_4655744548891
// MI455X (gfx1250) — matrix operand order repaired and checked by reading, not compiled, not run on hardware
//
#include <hip/hip_runtime.h>
#include <hip/hip_bf16.h>
#include <stdint.h>

// ---------------------------------------------------------------------------
// GPT-2 forward for MI455X (gfx1250): bf16 WMMA GEMMs + f32 accumulate,
// with double-buffered GLOBAL_LOAD_ASYNC_TO_LDS (ASYNCcnt) tile pipeline.
// ---------------------------------------------------------------------------

typedef __attribute__((ext_vector_type(16))) __bf16 v16bf;
typedef __attribute__((ext_vector_type(8)))  float  v8f;

#define DEV __device__ __forceinline__

DEV unsigned short f2bf(float f) {
  unsigned int u = __float_as_uint(f);
  if ((u & 0x7fffffffu) > 0x7f800000u) return (unsigned short)((u >> 16) | 0x0040u); // quiet NaN
  u += 0x7fffu + ((u >> 16) & 1u);   // round-to-nearest-even
  return (unsigned short)(u >> 16);
}

DEV float gelu_tanh(float x) {
  float x3 = x * x * x;
  return 0.5f * x * (1.0f + tanhf(0.7978845608028654f * (x + 0.044715f * x3)));
}

// Async 16-byte global -> LDS copy (per active lane). Tracked by ASYNCcnt.
// GV addressing: lds[vdst + ioffset] = mem[vaddr + ioffset].
DEV void async_b128(unsigned lds_off, const void* gptr) {
  asm volatile("global_load_async_to_lds_b128 %0, %1, off"
               :: "v"(lds_off), "v"((unsigned long long)(uintptr_t)gptr)
               : "memory");
}

// ---------------------------------------------------------------------------
// Embedding: x[b,s,:] = wte[id] + wpe[s]
// ---------------------------------------------------------------------------
__global__ void embed_kernel(const int* __restrict__ ids, const float* __restrict__ wte,
                             const float* __restrict__ wpe, float* __restrict__ x,
                             int S, int D) {
  int row = blockIdx.x;
  int s = row % S;
  long id = ids[row];
  const float* we = wte + id * (long)D;
  const float* pe = wpe + (long)s * D;
  float* o = x + (long)row * D;
  for (int d = threadIdx.x; d < D; d += blockDim.x) o[d] = we[d] + pe[d];
}

// ---------------------------------------------------------------------------
// LayerNorm (fp32 in) -> bf16 out
// ---------------------------------------------------------------------------
__global__ void ln_kernel(const float* __restrict__ x, const float* __restrict__ gamma,
                          const float* __restrict__ beta, unsigned short* __restrict__ out,
                          int D) {
  __shared__ float sh[256];
  int row = blockIdx.x, t = threadIdx.x;
  const float* xr = x + (long)row * D;
  float s = 0.f, s2 = 0.f;
  for (int d = t; d < D; d += 256) { float v = xr[d]; s += v; s2 += v * v; }
  sh[t] = s; __syncthreads();
  for (int o = 128; o > 0; o >>= 1) { if (t < o) sh[t] += sh[t + o]; __syncthreads(); }
  float mu = sh[0] / (float)D; __syncthreads();
  sh[t] = s2; __syncthreads();
  for (int o = 128; o > 0; o >>= 1) { if (t < o) sh[t] += sh[t + o]; __syncthreads(); }
  float var = sh[0] / (float)D - mu * mu;
  float rstd = rsqrtf(var + 1e-6f);
  unsigned short* orow = out + (long)row * D;
  for (int d = t; d < D; d += 256)
    orow[d] = f2bf((xr[d] - mu) * rstd * gamma[d] + beta[d]);
}

// ---------------------------------------------------------------------------
// QKV split: qkv fp32 [T,3D] -> Q [B,H,S,64], K^T [B,H,64,S], V [B,H,S,64] (bf16)
// ---------------------------------------------------------------------------
__global__ void split_kernel(const float* __restrict__ qkv, unsigned short* __restrict__ Q,
                             unsigned short* __restrict__ KT, unsigned short* __restrict__ Vt,
                             int S, int Hn) {
  const int D = 1024;
  int row = blockIdx.x;
  int b = row / S, s = row % S;
  const float* r = qkv + (long)row * 3 * D;
  for (int d = threadIdx.x; d < D; d += blockDim.x) {
    int h = d >> 6, hd = d & 63;
    long bh = (long)(b * Hn + h);
    Q [(bh * S + s) * 64 + hd]        = f2bf(r[d]);
    KT[(bh * 64 + hd) * (long)S + s]  = f2bf(r[D + d]);
    Vt[(bh * S + s) * 64 + hd]        = f2bf(r[2 * D + d]);
  }
}

// ---------------------------------------------------------------------------
// Causal softmax: scores fp32 [BH*S, S] -> probs bf16. Masked = -10000 (as ref).
// Assumes S == 1024 (4 cols/thread x 256 threads).
// ---------------------------------------------------------------------------
__global__ void softmax_kernel(const float* __restrict__ sc, unsigned short* __restrict__ pr,
                               int S) {
  __shared__ float sh[256];
  long row = blockIdx.x;
  int q = (int)(row % S);
  const float* r = sc + row * (long)S;
  unsigned short* o = pr + row * (long)S;
  int t = threadIdx.x;
  float v[4];
  float mx = -3.0e38f;
  #pragma unroll
  for (int k = 0; k < 4; k++) {
    int c = t + k * 256;
    float x = (c <= q) ? r[c] : -10000.0f;
    v[k] = x; mx = fmaxf(mx, x);
  }
  sh[t] = mx; __syncthreads();
  for (int o2 = 128; o2 > 0; o2 >>= 1) { if (t < o2) sh[t] = fmaxf(sh[t], sh[t + o2]); __syncthreads(); }
  mx = sh[0]; __syncthreads();
  float sum = 0.f;
  #pragma unroll
  for (int k = 0; k < 4; k++) { v[k] = __expf(v[k] - mx); sum += v[k]; }
  sh[t] = sum; __syncthreads();
  for (int o2 = 128; o2 > 0; o2 >>= 1) { if (t < o2) sh[t] += sh[t + o2]; __syncthreads(); }
  float inv = 1.0f / sh[0];
  #pragma unroll
  for (int k = 0; k < 4; k++) o[t + k * 256] = f2bf(v[k] * inv);
}

// ---------------------------------------------------------------------------
// fp32 -> bf16 weight staging with optional column padding (zeros in [N, Npad)).
// ---------------------------------------------------------------------------
__global__ void cvt_pad_kernel(const float* __restrict__ src, unsigned short* __restrict__ dst,
                               int N, int Npad, long total) {
  long i = (long)blockIdx.x * blockDim.x + threadIdx.x;
  long stride = (long)gridDim.x * blockDim.x;
  for (; i < total; i += stride) {
    long rr = i / Npad; int c = (int)(i % Npad);
    dst[i] = (c < N) ? f2bf(src[rr * (long)N + c]) : (unsigned short)0;
  }
}

// ---------------------------------------------------------------------------
// WMMA bf16 GEMM: C[M,N] = scale*(A[M,K] @ B[K,N]) (+bias)(+GELU)(+residual)
// A, B bf16 row-major; C fp32 or bf16. Batched via z -> (b = z/Hz, h = z%Hz).
// Tile: 128x128x32, 256 threads = 8 waves (4x2), wave tile 32x64 = 2x4 WMMAs.
// Tiles stream via double-buffered global_load_async_to_lds_b128:
//   - each thread issues exactly 4 async b128 copies per tile (2 A + 2 B),
//     so per-wave ASYNCcnt grows by a constant 4 per tile (A guard is uniform
//     since M % 128 == 0 for every call; B guard always has >=1 active lane
//     per wave), letting s_wait_asynccnt 0x4 retire exactly the current tile
//     while the next tile's DMA stays in flight behind the WMMAs.
//   - out-of-range B lanes (only the P@V GEMM, N=64 < BN, constant over K)
//     never issue loads; both LDS buffers are pre-zeroed once so their cells
//     contribute zeros to the WMMAs.
// Requires: K % 32 == 0, M % 128 == 0, Nld % 16 == 0 (true for all call sites).
// ---------------------------------------------------------------------------
template <bool BIAS, bool RESID, bool GELU_, bool OUTBF>
__global__ __launch_bounds__(256)
void gemm_wmma(int M, int N, int K,
               const unsigned short* __restrict__ A, long lda, long Ab, long Ah,
               const unsigned short* __restrict__ Bm, long ldb, long Bb, long Bh, int Nld,
               float* __restrict__ Cf, unsigned short* __restrict__ Cbf,
               long ldc, long Cb, long Ch,
               const float* __restrict__ bias, const float* __restrict__ resid,
               float scale, int Hz) {
  constexpr int BM = 128, BN = 128, BK = 32;
  __shared__ unsigned short lA[2][BM * BK];
  __shared__ unsigned short lB[2][BK * BN];

  int z = blockIdx.z, bz = z / Hz, hz = z % Hz;
  A  += (long)bz * Ab + (long)hz * Ah;
  Bm += (long)bz * Bb + (long)hz * Bh;
  long coff = (long)bz * Cb + (long)hz * Ch;

  int tid = threadIdx.x;
  int lane = tid & 31, wave = tid >> 5;
  int wm = wave >> 1, wn = wave & 1;                 // 4x2 wave grid
  int m0 = blockIdx.y * BM, n0 = blockIdx.x * BN;

  int ra = tid >> 1, ca = (tid & 1) * 16;            // A tile: 128 rows x 2x16-half chunks
  int kb = tid >> 3, nb = (tid & 7) * 16;            // B tile: 32 rows x 8x16-half chunks
  long gr = m0 + ra;
  int gc = n0 + nb;
  bool aok = (gr < M);
  bool bok = (gc + 16 <= Nld);

  // Pre-zero this thread's slots in both buffers (covers lanes with no loads).
  {
    uint4 zz = {0, 0, 0, 0};
    #pragma unroll
    for (int b2 = 0; b2 < 2; b2++) {
      *(uint4*)&lA[b2][ra * BK + ca]     = zz;
      *(uint4*)&lA[b2][ra * BK + ca + 8] = zz;
      *(uint4*)&lB[b2][kb * BN + nb]     = zz;
      *(uint4*)&lB[b2][kb * BN + nb + 8] = zz;
    }
  }
  __syncthreads();

  auto issue_tile = [&](int kt, int buf) {
    if (aok) {
      const unsigned short* ap = A + gr * lda + (kt + ca);
      async_b128((unsigned)(uintptr_t)&lA[buf][ra * BK + ca],     ap);
      async_b128((unsigned)(uintptr_t)&lA[buf][ra * BK + ca + 8], ap + 8);
    }
    if (bok) {
      const unsigned short* bp = Bm + (long)(kt + kb) * ldb + gc;
      async_b128((unsigned)(uintptr_t)&lB[buf][kb * BN + nb],     bp);
      async_b128((unsigned)(uintptr_t)&lB[buf][kb * BN + nb + 8], bp + 8);
    }
  };

  v8f acc[2][4] = {};
  int cur = 0;
  issue_tile(0, 0);

  for (int kt = 0; kt < K; kt += BK) {
    bool nxt = (kt + BK) < K;
    if (nxt) {
      issue_tile(kt + BK, cur ^ 1);
      asm volatile("s_wait_asynccnt 0x4" ::: "memory");   // current tile's 4 done
    } else {
      asm volatile("s_wait_asynccnt 0x0" ::: "memory");
    }
    __syncthreads();

    // Build fragments from LDS (A: contiguous b128 reads, B: strided 16-bit reads).
    // A 16x32 bf16: lanes 0-15 M=lane K=0..7/16..23; lanes 16-31 M=lane-16 K=8..15/24..31.
    union Frag { v16bf v; uint4 q[2]; };
    Frag af[2], bfr[4];
    int arow = (wm * 32 + (lane & 15)) * BK + (lane >> 4) * 8;
    af[0].q[0] = *(const uint4*)&lA[cur][arow];
    af[0].q[1] = *(const uint4*)&lA[cur][arow + 16];
    af[1].q[0] = *(const uint4*)&lA[cur][arow + 16 * BK];
    af[1].q[1] = *(const uint4*)&lA[cur][arow + 16 * BK + 16];
    // B 32x16 bf16: lanes 0-15 N=lane K=0..7/16..23; lanes 16-31 N=lane-16 K=8..15/24..31.
    int bcol = wn * 64 + (lane & 15);
    #pragma unroll
    for (int j = 0; j < 4; j++) {
      #pragma unroll
      for (int s = 0; s < 16; s++)
        bfr[j].v[s] = __builtin_bit_cast(__bf16, lB[cur][((lane >> 4) * 8 + (s & 7) + (s >> 3) * 16) * BN + bcol + j * 16]);
    }
    #pragma unroll
    for (int i = 0; i < 2; i++)
      #pragma unroll
      for (int j = 0; j < 4; j++)
        acc[i][j] = __builtin_amdgcn_wmma_f32_16x16x32_bf16(
            false, af[i].v, false, bfr[j].v, (short)0, acc[i][j], false, false);
    __syncthreads();   // everyone done reading buf[cur] before it is re-filled
    cur ^= 1;
  }

  // C/D 16x16 f32 layout: VGPR g -> row g (+8 for lanes 16-31), col = lane&15.
  int rbase = m0 + wm * 32 + ((lane >> 4) << 3);
  int cbase = n0 + wn * 64 + (lane & 15);
  #pragma unroll
  for (int i = 0; i < 2; i++) {
    #pragma unroll
    for (int j = 0; j < 4; j++) {
      int cj = cbase + j * 16;
      if (cj >= N) continue;
      #pragma unroll
      for (int g = 0; g < 8; g++) {
        int r = rbase + i * 16 + g;
        if (r >= M) continue;
        float v = acc[i][j][g] * scale;
        if (BIAS)  v += bias[cj];
        if (GELU_) v = gelu_tanh(v);
        long idx = coff + (long)r * ldc + cj;
        if (RESID) v += resid[idx];
        if (OUTBF) Cbf[idx] = f2bf(v);
        else       Cf[idx]  = v;
      }
    }
  }
}

// ---------------------------------------------------------------------------
// Host-side launch helpers
// ---------------------------------------------------------------------------
template <bool BIAS, bool RESID, bool GELU_, bool OUTBF>
static void launch_gemm(hipStream_t st, int M, int N, int K,
                        const unsigned short* A, long lda, long Ab, long Ah,
                        const unsigned short* Bm, long ldb, long Bb, long Bh, int Nld,
                        float* Cf, unsigned short* Cbf, long ldc, long Cb, long Ch,
                        const float* bias, const float* resid, float scale, int Z, int Hz) {
  dim3 g((N + 127) / 128, (M + 127) / 128, Z);
  gemm_wmma<BIAS, RESID, GELU_, OUTBF><<<g, 256, 0, st>>>(
      M, N, K, A, lda, Ab, Ah, Bm, ldb, Bb, Bh, Nld, Cf, Cbf, ldc, Cb, Ch,
      bias, resid, scale, Hz);
}

static void cvt_launch(const float* src, unsigned short* dst, int K, int N, int Npad,
                       hipStream_t st) {
  long total = (long)K * Npad;
  int blocks = (int)((total + 1023) / 1024);
  if (blocks > 8192) blocks = 8192;
  cvt_pad_kernel<<<blocks, 256, 0, st>>>(src, dst, N, Npad, total);
}

// ---------------------------------------------------------------------------
// kernel_launch: full GPT-2 forward.
// d_in flattening (JAX pytree, dict keys sorted):
//   0: input_ids
//   1 + l*12 + {0:attn_proj.b, 1:attn_proj.w, 2:c_attn.b, 3:c_attn.w,
//               4:c_fc.b, 5:c_fc.w, 6:ln_1.bias, 7:ln_1.scale,
//               8:ln_2.bias, 9:ln_2.scale, 10:mlp_proj.b, 11:mlp_proj.w}
//   49: lm_head.w, 50: ln_f.bias, 51: ln_f.scale, 52: wpe, 53: wte
// ---------------------------------------------------------------------------
extern "C" void kernel_launch(void* const* d_in, const int* in_sizes, int n_in,
                              void* d_out, int out_size, void* d_ws, size_t ws_size,
                              hipStream_t stream) {
  (void)in_sizes; (void)n_in; (void)out_size; (void)ws_size;
  const int Bn = 2, S = 1024, D = 1024, Hn = 16, Dff = 4096;
  const int Vv = 50257, Vp = 50304;  // Vp = V rounded up to 128 (keeps LDS loads aligned)
  const int T = Bn * S, Ln = 4;

  const int*   ids   = (const int*)  d_in[0];
  const float* lm_w  = (const float*)d_in[49];
  const float* lnf_b = (const float*)d_in[50];
  const float* lnf_s = (const float*)d_in[51];
  const float* wpe   = (const float*)d_in[52];
  const float* wte   = (const float*)d_in[53];

  char* p = (char*)d_ws;
  auto alloc = [&](size_t bytes) -> void* {
    void* r = (void*)p;
    p += (bytes + 255) & ~(size_t)255;
    return r;
  };
  float*          x   = (float*)         alloc((size_t)T * D * 4);        // residual stream
  unsigned short* lnb = (unsigned short*)alloc((size_t)T * D * 2);        // LN output (bf16)
  float*          qkv = (float*)         alloc((size_t)T * 3 * D * 4);
  unsigned short* Q   = (unsigned short*)alloc((size_t)T * D * 2);
  unsigned short* KT  = (unsigned short*)alloc((size_t)T * D * 2);
  unsigned short* Vt  = (unsigned short*)alloc((size_t)T * D * 2);
  float*          sc  = (float*)         alloc((size_t)Bn * Hn * S * S * 4);  // 128 MB, fits L2
  unsigned short* pr  = (unsigned short*)alloc((size_t)Bn * Hn * S * S * 2);
  unsigned short* aO  = (unsigned short*)alloc((size_t)T * D * 2);
  unsigned short* hb  = (unsigned short*)alloc((size_t)T * Dff * 2);
  unsigned short* wb  = (unsigned short*)alloc((size_t)D * Vp * 2);       // bf16 weight staging

  embed_kernel<<<T, 256, 0, stream>>>(ids, wte, wpe, x, S, D);

  for (int l = 0; l < Ln; l++) {
    const float* apb = (const float*)d_in[1 + l * 12 + 0];
    const float* apw = (const float*)d_in[1 + l * 12 + 1];
    const float* cab = (const float*)d_in[1 + l * 12 + 2];
    const float* caw = (const float*)d_in[1 + l * 12 + 3];
    const float* cfb = (const float*)d_in[1 + l * 12 + 4];
    const float* cfw = (const float*)d_in[1 + l * 12 + 5];
    const float* l1b = (const float*)d_in[1 + l * 12 + 6];
    const float* l1s = (const float*)d_in[1 + l * 12 + 7];
    const float* l2b = (const float*)d_in[1 + l * 12 + 8];
    const float* l2s = (const float*)d_in[1 + l * 12 + 9];
    const float* mpb = (const float*)d_in[1 + l * 12 + 10];
    const float* mpw = (const float*)d_in[1 + l * 12 + 11];

    // x -> ln1 -> qkv = ln1 @ Wqkv + b
    ln_kernel<<<T, 256, 0, stream>>>(x, l1s, l1b, lnb, D);
    cvt_launch(caw, wb, D, 3 * D, 3 * D, stream);
    launch_gemm<true, false, false, false>(stream, T, 3 * D, D,
        lnb, D, 0, 0, wb, 3 * D, 0, 0, 3 * D,
        qkv, nullptr, 3 * D, 0, 0, cab, nullptr, 1.0f, 1, 1);

    split_kernel<<<T, 256, 0, stream>>>(qkv, Q, KT, Vt, S, Hn);

    // scores = (Q @ K^T) / sqrt(64), per (b,h)
    launch_gemm<false, false, false, false>(stream, S, S, 64,
        Q,  64, (long)Hn * S * 64, (long)S * 64,
        KT, S,  (long)Hn * 64 * S, (long)64 * S, S,
        sc, nullptr, S, (long)Hn * S * S, (long)S * S,
        nullptr, nullptr, 0.125f, Bn * Hn, Hn);

    softmax_kernel<<<Bn * Hn * S, 256, 0, stream>>>(sc, pr, S);

    // O = P @ V, written merged-head into aO[B,S,D] (bf16)
    launch_gemm<false, false, false, true>(stream, S, 64, S,
        pr, S,  (long)Hn * S * S, (long)S * S,
        Vt, 64, (long)Hn * S * 64, (long)S * 64, 64,
        nullptr, aO, D, (long)S * D, 64,
        nullptr, nullptr, 1.0f, Bn * Hn, Hn);

    // x = x + O @ Wproj + b
    cvt_launch(apw, wb, D, D, D, stream);
    launch_gemm<true, true, false, false>(stream, T, D, D,
        aO, D, 0, 0, wb, D, 0, 0, D,
        x, nullptr, D, 0, 0, apb, x, 1.0f, 1, 1);

    // h = gelu(ln2(x) @ Wfc + b) (bf16)
    ln_kernel<<<T, 256, 0, stream>>>(x, l2s, l2b, lnb, D);
    cvt_launch(cfw, wb, D, Dff, Dff, stream);
    launch_gemm<true, false, true, true>(stream, T, Dff, D,
        lnb, D, 0, 0, wb, Dff, 0, 0, Dff,
        nullptr, hb, Dff, 0, 0, cfb, nullptr, 1.0f, 1, 1);

    // x = x + h @ Wmlp + b
    cvt_launch(mpw, wb, Dff, D, D, stream);
    launch_gemm<true, true, false, false>(stream, T, D, Dff,
        hb, Dff, 0, 0, wb, D, 0, 0, D,
        x, nullptr, D, 0, 0, mpb, x, 1.0f, 1, 1);
  }

  // logits = ln_f(x) @ Wlm  (N padded to 50304 in staging, store-guarded to 50257)
  ln_kernel<<<T, 256, 0, stream>>>(x, lnf_s, lnf_b, lnb, D);
  cvt_launch(lm_w, wb, D, Vv, Vp, stream);
  launch_gemm<false, false, false, false>(stream, T, Vv, D,
      lnb, D, 0, 0, wb, Vp, 0, 0, Vp,
      (float*)d_out, nullptr, Vv, 0, 0, nullptr, nullptr, 1.0f, 1, 1);
}
